// KTBM_mat_nopers_37323265802340
// MI455X (gfx1250) — matrix operand order repaired and checked by reading, not compiled, not run on hardware
//
#include <hip/hip_runtime.h>
#include <hip/hip_bf16.h>

// ---------------------------------------------------------------------------
// MI455X (gfx1250, wave32) implementation of the recurrent-memory scan.
// 16 blocks x 256 threads; each block owns 16 batch rows for all 512 steps.
// All matmuls on the critical path are v_wmma_f32_16x16x32_f16 (M=16 tiles).
// ---------------------------------------------------------------------------

typedef __attribute__((ext_vector_type(16))) _Float16 v16h;
typedef __attribute__((ext_vector_type(8)))  _Float16 v8h;
typedef __attribute__((ext_vector_type(8)))  float    v8f;

#define B_   256
#define S_   512
#define C_   64
#define DV_  128
#define BH_  128

// ---------------- WMMA fragment helpers (ISA 7.12.2 layouts) ----------------

__device__ __forceinline__ v8f vzero8() {
  v8f z;
#pragma unroll
  for (int e = 0; e < 8; ++e) z[e] = 0.0f;
  return z;
}

// A fragment 16x32 f16 from LDS (row-major, leading dim `ld` halves).
// lane<16: halves 0..7 = K0..7, 8..15 = K16..23 ; lane>=16: +8.
__device__ __forceinline__ v16h load_a(const _Float16* As, int ld, int kt, int lane) {
  int m  = lane & 15;
  int kb = (lane >> 4) << 3;
  const _Float16* p = As + m * ld + kt * 32;
  v8h lo = *(const v8h*)(p + kb);
  v8h hi = *(const v8h*)(p + 16 + kb);
  v16h r;
#pragma unroll
  for (int e = 0; e < 8; ++e) { r[e] = lo[e]; r[8 + e] = hi[e]; }
  return r;
}

// B fragment 32x16 f16 from packed global weights.
// Packed so each lane's 16 halves are contiguous (two b128 loads).
__device__ __forceinline__ v16h load_b(const _Float16* Wp, int NT, int kt, int nt, int lane) {
  const _Float16* p = Wp + ((((size_t)(kt * NT + nt) << 5) + lane) << 4);
  v8h lo = *(const v8h*)p;
  v8h hi = *(const v8h*)(p + 8);
  v16h r;
#pragma unroll
  for (int e = 0; e < 8; ++e) { r[e] = lo[e]; r[8 + e] = hi[e]; }
  return r;
}

__device__ __forceinline__ v8f wmma16(v16h a, v16h b, v8f c) {
  return __builtin_amdgcn_wmma_f32_16x16x32_f16(false, a, false, b, (short)0, c, false, false);
}

// C += A(16 x 32*kTiles) @ W(32*kTiles x 16*NT) for one N-tile `nt`.
__device__ __forceinline__ v8f mm(const _Float16* As, int ld, const _Float16* Wp,
                                  int NT, int nt, int kTiles, int lane, v8f acc) {
  for (int kt = 0; kt < kTiles; ++kt) {
    v16h a = load_a(As, ld, kt, lane);
    v16h b = load_b(Wp, NT, kt, nt, lane);
    acc = wmma16(a, b, acc);
  }
  return acc;
}

__device__ __forceinline__ float sigm(float x) { return 1.0f / (1.0f + __expf(-x)); }

// ---------------------------- argument bundle -------------------------------

struct ScanArgs {
  const int *q_data, *a_data, *l_data, *d_data;
  const float *q_embed, *l_embed, *q_corr, *l_corr;
  const float *a_embed, *d_embed, *q_b_embed, *l_b_embed;
  const float *bmq, *bml, *be_q, *be_l, *ba_q, *ba_l, *bsum1, *bpref;
  const float *b_i, *b_ik, *b_g, *b_gk, *b_f, *b_fk, *b_o, *b_ok;
  const float *Wsum2, *bsum2, *Wbt, *bbt;
  const _Float16 *pWmq, *pWml, *pWeq, *pWel, *pWaq, *pWal, *pWebh, *pWabh;
  const _Float16 *pWs1a, *pWs1b, *pWs1c, *pWprefh, *pWqal;
  const _Float16 *pWg0, *pWg1, *pWg2, *pWg3;   // [W_g; W_gk; W_gh] 384x128, gate order i,g,f,o
  float* V;     // [B][64][128] f32 state
  float* out;   // [B][S][2]
};

// ------------------------------- prep kernels -------------------------------

// Pack f32 KxN (row-major) weight into f16 B-fragment layout.
__global__ void pack_w(const float* __restrict__ W, _Float16* __restrict__ out, int K, int N) {
  int idx = blockIdx.x * 256 + threadIdx.x;
  if (idx >= K * N) return;
  int e = idx & 15, lane = (idx >> 4) & 31, tile = idx >> 9;
  int NT = N >> 4;
  int nt = tile % NT, kt = tile / NT;
  int k = kt * 32 + ((lane >> 4) << 3) + (e & 7) + ((e >> 3) << 4);
  int n = nt * 16 + (lane & 15);
  out[idx] = (_Float16)W[k * N + n];
}

// Pack concat([W0;W1;W2]) with 128 rows each (K=384), N=128.
__global__ void pack_cat3(const float* __restrict__ W0, const float* __restrict__ W1,
                          const float* __restrict__ W2, _Float16* __restrict__ out, int N) {
  const int K = 384;
  int idx = blockIdx.x * 256 + threadIdx.x;
  if (idx >= K * N) return;
  int e = idx & 15, lane = (idx >> 4) & 31, tile = idx >> 9;
  int NT = N >> 4;
  int nt = tile % NT, kt = tile / NT;
  int k = kt * 32 + ((lane >> 4) << 3) + (e & 7) + ((e >> 3) << 4);
  int n = nt * 16 + (lane & 15);
  float v;
  if (k < 128)      v = W0[k * N + n];
  else if (k < 256) v = W1[(k - 128) * N + n];
  else              v = W2[(k - 256) * N + n];
  out[idx] = (_Float16)v;
}

__global__ void vinit(const float* __restrict__ V0, float* __restrict__ V) {
  int idx = blockIdx.x * 256 + threadIdx.x;
  if (idx < B_ * C_ * DV_) V[idx] = V0[idx & (C_ * DV_ - 1)];
}

// ------------------------------- scan kernel --------------------------------

__global__ __launch_bounds__(256) void scan_kernel(ScanArgs args) {
  const int tid  = threadIdx.x;
  const int lane = tid & 31;
  const int wav  = tid >> 5;          // 8 waves
  const int bo   = blockIdx.x * 16;   // 16 batch rows per block

  // f16 A operands (16-byte aligned for b128 LDS loads)
  __shared__ __align__(16) _Float16 sAgate[16 * 384];  // [x | rc | h]
  __shared__ __align__(16) _Float16 sAh  [16 * 128];   // h_new f16
  __shared__ __align__(16) _Float16 sArcq[16 * 128];   // rc_q f16
  __shared__ __align__(16) _Float16 sAqa [16 * 128];   // [qe|ae]
  __shared__ __align__(16) _Float16 sAle [16 * 64];
  __shared__ __align__(16) _Float16 sAxq [16 * 128];   // [qb|de]
  __shared__ __align__(16) _Float16 sAxl [16 * 128];   // [lb|de]
  __shared__ __align__(16) _Float16 sAqal[16 * 192];   // [qb|ae|lb]
  __shared__ __align__(16) _Float16 sAqe [16 * 64];
  // f32 working arrays
  __shared__ float sCq[16 * 64], sCl[16 * 64];
  __shared__ float sEpre[16 * 128], sAdpre[16 * 128];
  __shared__ float sSum[16 * 128], sPref[16 * 64];
  __shared__ float sC[16 * 128];                        // cell state
  __shared__ int   sMd[16];

  for (int i = tid; i < 16 * 128; i += 256) { sC[i] = 0.0f; sAh[i] = (_Float16)0.0f; }
  __syncthreads();

  for (int t = 0; t < S_; ++t) {
    // ---- Phase 0: gather embeddings into f16 A-operands ----
    {
      int row = tid >> 4, sub = tid & 15;
      int b   = bo + row;
      int qi  = args.q_data[b * S_ + t];
      int li  = args.l_data[b * S_ + t];
      int ai  = args.a_data[b * S_ + t];
      int di  = args.d_data[b * S_ + t];
#pragma unroll
      for (int j = 0; j < 4; ++j) {
        int c = sub * 4 + j;
        float qe = args.q_embed  [qi * 64 + c];
        float le = args.l_embed  [li * 64 + c];
        float ae = args.a_embed  [ai * 64 + c];
        float de = args.d_embed  [di * 64 + c];
        float qb = args.q_b_embed[qi * 64 + c];
        float lb = args.l_b_embed[li * 64 + c];
        sAqa [row * 128 + c]       = (_Float16)qe;
        sAqa [row * 128 + 64 + c]  = (_Float16)ae;
        sAle [row * 64 + c]        = (_Float16)le;
        sAxq [row * 128 + c]       = (_Float16)qb;
        sAxq [row * 128 + 64 + c]  = (_Float16)de;
        sAxl [row * 128 + c]       = (_Float16)lb;
        sAxl [row * 128 + 64 + c]  = (_Float16)de;
        sAqal[row * 192 + c]       = (_Float16)qb;
        sAqal[row * 192 + 64 + c]  = (_Float16)ae;
        sAqal[row * 192 + 128 + c] = (_Float16)lb;
        sAqe [row * 64 + c]        = (_Float16)qe;
        sCq  [row * 64 + c] = args.q_corr[qi * 64 + c];
        sCl  [row * 64 + c] = args.l_corr[li * 64 + c];
      }
#pragma unroll
      for (int j = 0; j < 8; ++j) {           // refresh h slot of gate operand
        int col = sub * 8 + j;
        sAgate[row * 384 + 256 + col] = sAh[row * 128 + col];
      }
      if (sub == 0) sMd[row] = (di == 0);
    }
    __syncthreads();

    // ---- Phase 1: softmax of correlation rows (one thread per row) ----
    if (tid < 16) {
      int row = tid;
      float mx = -1e30f;
      for (int c = 0; c < 64; ++c) mx = fmaxf(mx, sCq[row * 64 + c]);
      float sm = 0.0f;
      for (int c = 0; c < 64; ++c) { float e = __expf(sCq[row * 64 + c] - mx); sCq[row * 64 + c] = e; sm += e; }
      float inv = 1.0f / sm;
      for (int c = 0; c < 64; ++c) sCq[row * 64 + c] *= inv;
      mx = -1e30f;
      for (int c = 0; c < 64; ++c) mx = fmaxf(mx, sCl[row * 64 + c]);
      sm = 0.0f;
      for (int c = 0; c < 64; ++c) { float e = __expf(sCl[row * 64 + c] - mx); sCl[row * 64 + c] = e; sm += e; }
      inv = 1.0f / sm;
      for (int c = 0; c < 64; ++c) sCl[row * 64 + c] *= inv;
    }
    __syncthreads();

    // ---- Phase 2a: rc = corr . V  (per-batch GEMV, f32 VALU, V from L2) ----
    {
      int b16 = tid >> 4, v0 = (tid & 15) * 8;
      float rq[8], rl[8];
#pragma unroll
      for (int j = 0; j < 8; ++j) { rq[j] = 0.0f; rl[j] = 0.0f; }
      const float* Vb = args.V + ((size_t)(bo + b16) * C_) * DV_;
      for (int c = 0; c < C_; ++c) {
        float wq = sCq[b16 * 64 + c], wl = sCl[b16 * 64 + c];
        const float* vp = Vb + c * DV_ + v0;
#pragma unroll
        for (int j = 0; j < 8; ++j) { float vv = vp[j]; rq[j] += wq * vv; rl[j] += wl * vv; }
      }
      bool mq = sMd[b16] != 0;
#pragma unroll
      for (int j = 0; j < 8; ++j) {
        sAgate[b16 * 384 + 128 + v0 + j] = (_Float16)(mq ? rq[j] : rl[j]);
        sArcq [b16 * 128 + v0 + j]       = (_Float16)rq[j];
      }
    }

    // ---- Phase 2b: input-only WMMAs (x, e/ad pre, summ pre, pref pre) ----
    for (int job = wav; job < 28; job += 8) {
      if (job < 8) {                       // x = select([qb,de]@Wmq, [lb,de]@Wml)
        int nt = job;
        v8f aq = mm(sAxq, 128, args.pWmq, 8, nt, 4, lane, vzero8());
        v8f al = mm(sAxl, 128, args.pWml, 8, nt, 4, lane, vzero8());
        int n = nt * 16 + (lane & 15), rb = (lane >> 4) << 3;
        float bq = args.bmq[n], bl = args.bml[n];
#pragma unroll
        for (int r = 0; r < 8; ++r) {
          int row = rb + r;
          float x = sMd[row] ? (aq[r] + bq) : (al[r] + bl);
          sAgate[row * 384 + n] = (_Float16)x;
        }
      } else if (job < 16) {               // e/ad pre-activations (both branches)
        int nt = job - 8;
        v8f eq  = mm(sAqa, 128, args.pWeq, 8, nt, 4, lane, vzero8());
        v8f el  = mm(sAle,  64, args.pWel, 8, nt, 2, lane, vzero8());
        v8f aaq = mm(sAqa, 128, args.pWaq, 8, nt, 4, lane, vzero8());
        v8f aal = mm(sAle,  64, args.pWal, 8, nt, 2, lane, vzero8());
        int n = nt * 16 + (lane & 15), rb = (lane >> 4) << 3;
        float beq = args.be_q[n], bel = args.be_l[n];
        float baq = args.ba_q[n], bal = args.ba_l[n];
#pragma unroll
        for (int r = 0; r < 8; ++r) {
          int row = rb + r;
          if (sMd[row]) { sEpre[row * 128 + n] = eq[r] + beq; sAdpre[row * 128 + n] = aaq[r] + baq; }
          else          { sEpre[row * 128 + n] = el[r] + bel; sAdpre[row * 128 + n] = aal[r] + bal; }
        }
      } else if (job < 24) {               // summ pre: qe @ Wsum1[0:64]
        int nt = job - 16;
        v8f a = mm(sAqe, 64, args.pWs1a, 8, nt, 2, lane, vzero8());
        int n = nt * 16 + (lane & 15), rb = (lane >> 4) << 3;
        float bs = args.bsum1[n];
#pragma unroll
        for (int r = 0; r < 8; ++r) sSum[(rb + r) * 128 + n] = a[r] + bs;
      } else {                             // pref pre: [qb,ae,lb] @ Wpref[128:320]
        int nt = job - 24;
        v8f a = mm(sAqal, 192, args.pWqal, 4, nt, 6, lane, vzero8());
        int n = nt * 16 + (lane & 15), rb = (lane >> 4) << 3;
        float bp = args.bpref[n];
#pragma unroll
        for (int r = 0; r < 8; ++r) sPref[(rb + r) * 64 + n] = a[r] + bp;
      }
    }
    __syncthreads();

    // ---- Phase 3: fused 4-gate WMMA + LSTM cell update (register C-frags) --
    {
      int nt = wav;                                 // 8 waves -> 8 N-tiles
      int n = nt * 16 + (lane & 15), rb = (lane >> 4) << 3;
      const _Float16* Wg[4] = { args.pWg0, args.pWg1, args.pWg2, args.pWg3 };
      v8f acc[4];
      float bi = args.b_i[n] + args.b_ik[n];
      float bg = args.b_g[n] + args.b_gk[n];
      float bf = args.b_f[n] + args.b_fk[n];
      float bo_ = args.b_o[n] + args.b_ok[n];
#pragma unroll
      for (int r = 0; r < 8; ++r) { acc[0][r] = bi; acc[1][r] = bg; acc[2][r] = bf; acc[3][r] = bo_; }
      for (int kt = 0; kt < 12; ++kt) {             // K = 384 = [x|rc|h]
        v16h a = load_a(sAgate, 384, kt, lane);
#pragma unroll
        for (int g = 0; g < 4; ++g)
          acc[g] = wmma16(a, load_b(Wg[g], 8, kt, nt, lane), acc[g]);
      }
#pragma unroll
      for (int r = 0; r < 8; ++r) {
        int row = rb + r;
        float iv = sigm(acc[0][r]);
        float gv = tanhf(acc[1][r]);
        float fv = sigm(acc[2][r]);
        float ov = sigm(acc[3][r]);
        float cn = fv * sC[row * 128 + n] + iv * gv;
        sC[row * 128 + n] = cn;
        sAh[row * 128 + n] = (_Float16)(ov * tanhf(cn));
      }
    }
    __syncthreads();

    // ---- Phase 4: h_new-dependent WMMAs: e, ad, summ, pref ----
    for (int job = wav; job < 28; job += 8) {
      if (job < 8) {
        int nt = job;
        v8f a = mm(sAh, 128, args.pWebh, 8, nt, 4, lane, vzero8());
        int n = nt * 16 + (lane & 15), rb = (lane >> 4) << 3;
#pragma unroll
        for (int r = 0; r < 8; ++r) {
          int i = (rb + r) * 128 + n;
          sEpre[i] = sigm(sEpre[i] + a[r]);
        }
      } else if (job < 16) {
        int nt = job - 8;
        v8f a = mm(sAh, 128, args.pWabh, 8, nt, 4, lane, vzero8());
        int n = nt * 16 + (lane & 15), rb = (lane >> 4) << 3;
#pragma unroll
        for (int r = 0; r < 8; ++r) {
          int i = (rb + r) * 128 + n;
          sAdpre[i] = tanhf(sAdpre[i] + a[r]);
        }
      } else if (job < 24) {
        int nt = job - 16;
        v8f a = mm(sArcq, 128, args.pWs1b, 8, nt, 4, lane, vzero8());
        a     = mm(sAh,   128, args.pWs1c, 8, nt, 4, lane, a);
        int n = nt * 16 + (lane & 15), rb = (lane >> 4) << 3;
#pragma unroll
        for (int r = 0; r < 8; ++r) {
          int i = (rb + r) * 128 + n;
          sSum[i] = tanhf(sSum[i] + a[r]);
        }
      } else {
        int nt = job - 24;
        v8f a = mm(sAh, 128, args.pWprefh, 4, nt, 4, lane, vzero8());
        int n = nt * 16 + (lane & 15), rb = (lane >> 4) << 3;
#pragma unroll
        for (int r = 0; r < 8; ++r) {
          int i = (rb + r) * 64 + n;
          sPref[i] = tanhf(sPref[i] + a[r]);
        }
      }
    }
    __syncthreads();

    // ---- Phase 5: rank-1 V update (f32 VALU) + outputs ----
    {
      int b16 = tid >> 4, v0 = (tid & 15) * 8;
      bool mq = sMd[b16] != 0;
      float ev[8], av[8];
#pragma unroll
      for (int j = 0; j < 8; ++j) {
        ev[j] = sEpre [b16 * 128 + v0 + j];
        av[j] = sAdpre[b16 * 128 + v0 + j];
      }
      float* Vb = args.V + ((size_t)(bo + b16) * C_) * DV_ + v0;
      for (int c = 0; c < C_; ++c) {
        float w = mq ? sCq[b16 * 64 + c] : sCl[b16 * 64 + c];
        float* vp = Vb + c * DV_;
#pragma unroll
        for (int j = 0; j < 8; ++j) {
          float vv = vp[j];
          vp[j] = vv * (1.0f - w * ev[j]) + w * av[j];
        }
      }
    }
    if (tid < 32) {
      int b16 = tid & 15, bg = bo + b16;
      if (tid < 16) {
        float s = args.bsum2[0];
        for (int k = 0; k < 128; ++k) s += sSum[b16 * 128 + k] * args.Wsum2[k];
        args.out[((size_t)bg * S_ + t) * 2 + 0] = sigm(s);
      } else {
        float s = args.bbt[0];
        for (int k = 0; k < 64; ++k) s += sPref[b16 * 64 + k] * args.Wbt[k];
        args.out[((size_t)bg * S_ + t) * 2 + 1] = sigm(s);
      }
    }
    __syncthreads();
  }
}

// ------------------------------ host launcher -------------------------------

extern "C" void kernel_launch(void* const* d_in, const int* in_sizes, int n_in,
                              void* d_out, int out_size, void* d_ws, size_t ws_size,
                              hipStream_t stream) {
  auto F = [&](int i) { return (const float*)d_in[i]; };
  const int* q_data = (const int*)d_in[0];
  const int* a_data = (const int*)d_in[1];
  const int* l_data = (const int*)d_in[2];
  const int* d_data = (const int*)d_in[3];
  // d_in[4] = s_data (unused)
  // params (dict order): 5 q_embed, 6 l_embed, 7 q_corr, 8 l_corr, 9 a_embed,
  // 10 d_embed, 11 q_b_embed, 12 l_b_embed, 13 V0, 14 We_q, 15 be_q, 16 We_l,
  // 17 be_l, 18 We_bh, 19 Wa_q, 20 ba_q, 21 Wa_l, 22 ba_l, 23 Wa_bh, 24 Wmq,
  // 25 bmq, 26 Wml, 27 bml, 28 Wsum1, 29 bsum1, 30 Wsum2, 31 bsum2, 32 Wpref,
  // 33 bpref, 34 Wbt, 35 bbt, then per gate g in (i,g,f,o):
  // W_g, b_g, W_gk, b_gk, W_gh -> 36..55

  char* ws = (char*)d_ws;
  size_t off = 0;
  float* V = (float*)ws;
  off += (size_t)B_ * C_ * DV_ * sizeof(float);
  auto alloc_h = [&](size_t nHalves) {
    off = (off + 31) & ~(size_t)31;
    _Float16* p = (_Float16*)(ws + off);
    off += nHalves * sizeof(_Float16);
    return p;
  };
  _Float16* pWmq   = alloc_h(128 * 128);
  _Float16* pWml   = alloc_h(128 * 128);
  _Float16* pWg0   = alloc_h(384 * 128);
  _Float16* pWg1   = alloc_h(384 * 128);
  _Float16* pWg2   = alloc_h(384 * 128);
  _Float16* pWg3   = alloc_h(384 * 128);
  _Float16* pWeq   = alloc_h(128 * 128);
  _Float16* pWel   = alloc_h( 64 * 128);
  _Float16* pWaq   = alloc_h(128 * 128);
  _Float16* pWal   = alloc_h( 64 * 128);
  _Float16* pWebh  = alloc_h(128 * 128);
  _Float16* pWabh  = alloc_h(128 * 128);
  _Float16* pWs1a  = alloc_h( 64 * 128);
  _Float16* pWs1b  = alloc_h(128 * 128);
  _Float16* pWs1c  = alloc_h(128 * 128);
  _Float16* pWprefh= alloc_h(128 *  64);
  _Float16* pWqal  = alloc_h(192 *  64);

  auto gridFor = [](int n) { return dim3((n + 255) / 256); };

  pack_w<<<gridFor(128 * 128), 256, 0, stream>>>(F(24), pWmq, 128, 128);
  pack_w<<<gridFor(128 * 128), 256, 0, stream>>>(F(26), pWml, 128, 128);
  pack_cat3<<<gridFor(384 * 128), 256, 0, stream>>>(F(36), F(38), F(40), pWg0, 128);
  pack_cat3<<<gridFor(384 * 128), 256, 0, stream>>>(F(41), F(43), F(45), pWg1, 128);
  pack_cat3<<<gridFor(384 * 128), 256, 0, stream>>>(F(46), F(48), F(50), pWg2, 128);
  pack_cat3<<<gridFor(384 * 128), 256, 0, stream>>>(F(51), F(53), F(55), pWg3, 128);
  pack_w<<<gridFor(128 * 128), 256, 0, stream>>>(F(14), pWeq, 128, 128);
  pack_w<<<gridFor( 64 * 128), 256, 0, stream>>>(F(16), pWel,  64, 128);
  pack_w<<<gridFor(128 * 128), 256, 0, stream>>>(F(19), pWaq, 128, 128);
  pack_w<<<gridFor( 64 * 128), 256, 0, stream>>>(F(21), pWal,  64, 128);
  pack_w<<<gridFor(128 * 128), 256, 0, stream>>>(F(18), pWebh, 128, 128);
  pack_w<<<gridFor(128 * 128), 256, 0, stream>>>(F(23), pWabh, 128, 128);
  pack_w<<<gridFor( 64 * 128), 256, 0, stream>>>(F(28),             pWs1a,  64, 128);
  pack_w<<<gridFor(128 * 128), 256, 0, stream>>>(F(28) +  64 * 128, pWs1b, 128, 128);
  pack_w<<<gridFor(128 * 128), 256, 0, stream>>>(F(28) + 192 * 128, pWs1c, 128, 128);
  pack_w<<<gridFor(128 *  64), 256, 0, stream>>>(F(32),             pWprefh, 128, 64);
  pack_w<<<gridFor(192 *  64), 256, 0, stream>>>(F(32) + 128 * 64,  pWqal,  192, 64);
  vinit<<<gridFor(B_ * C_ * DV_), 256, 0, stream>>>(F(13), V);

  ScanArgs a;
  a.q_data = q_data; a.a_data = a_data; a.l_data = l_data; a.d_data = d_data;
  a.q_embed = F(5); a.l_embed = F(6); a.q_corr = F(7); a.l_corr = F(8);
  a.a_embed = F(9); a.d_embed = F(10); a.q_b_embed = F(11); a.l_b_embed = F(12);
  a.bmq = F(25); a.bml = F(27); a.be_q = F(15); a.be_l = F(17);
  a.ba_q = F(20); a.ba_l = F(22); a.bsum1 = F(29); a.bpref = F(33);
  a.b_i = F(37); a.b_ik = F(39); a.b_g = F(42); a.b_gk = F(44);
  a.b_f = F(47); a.b_fk = F(49); a.b_o = F(52); a.b_ok = F(54);
  a.Wsum2 = F(30); a.bsum2 = F(31); a.Wbt = F(34); a.bbt = F(35);
  a.pWmq = pWmq; a.pWml = pWml; a.pWeq = pWeq; a.pWel = pWel;
  a.pWaq = pWaq; a.pWal = pWal; a.pWebh = pWebh; a.pWabh = pWabh;
  a.pWs1a = pWs1a; a.pWs1b = pWs1b; a.pWs1c = pWs1c;
  a.pWprefh = pWprefh; a.pWqal = pWqal;
  a.pWg0 = pWg0; a.pWg1 = pWg1; a.pWg2 = pWg2; a.pWg3 = pWg3;
  a.V = V; a.out = (float*)d_out;

  scan_kernel<<<dim3(B_ / 16), dim3(256), 0, stream>>>(a);
}
